// ConvLSTMBlock_v3_32461362823770
// MI455X (gfx1250) — matrix operand order repaired and checked by reading, not compiled, not run on hardware
//
#include <hip/hip_runtime.h>

typedef _Float16 h8   __attribute__((ext_vector_type(8)));
typedef _Float16 v16h __attribute__((ext_vector_type(16)));
typedef float    v8f  __attribute__((ext_vector_type(8)));

#define BB   4
#define TT   16
#define HH   64
#define WW   64
#define HP   66         // padded (1-pixel zero halo)
#define WP   66
#define CIN  32
#define COUT 64
#define G4   256        // 4*Cout
#define KTOT 864        // 3*3*32 + 3*3*64
#define LDSJ 872        // per-j row stride (halfs), 16B-aligned
#define LDSG 3496       // per-n16 group stride (halfs) = 4*872+8: 16 distinct bank starts
#define WMMA_F16(A, B, C) \
    __builtin_amdgcn_wmma_f32_16x16x32_f16(false, (A), false, (B), (short)0, (C), false, false)

// ---------------- prep kernels ----------------

// Convert x (B,T,H,W,CIN) f32 -> padded f16 (B,T,HP,WP,CIN) with zero halo.
__global__ void pad_x_f16(const float* __restrict__ src, _Float16* __restrict__ dst) {
    int i = blockIdx.x * 256 + threadIdx.x;
    const int total = BB * TT * HP * WP * CIN;
    if (i >= total) return;
    int c  = i % CIN;  int r = i / CIN;
    int px = r % WP;   r /= WP;
    int py = r % HP;   r /= HP;            // r = b*TT + t
    _Float16 v = (_Float16)0.0f;
    if (py >= 1 && py <= HH && px >= 1 && px <= WW)
        v = (_Float16)src[(((size_t)r * HH + (py - 1)) * WW + (px - 1)) * CIN + c];
    dst[i] = v;
}

// wp[n][k], k = 0..287 from input kernel (tap*32+ci), k = 288..863 from rec kernel (tap*64+ci)
__global__ void pack_weights(const float* __restrict__ kin, const float* __restrict__ krec,
                             _Float16* __restrict__ wp) {
    int i = blockIdx.x * 256 + threadIdx.x;
    if (i >= G4 * KTOT) return;
    int n = i / KTOT;
    int k = i % KTOT;
    float v;
    if (k < 288) v = kin[(size_t)k * G4 + n];            // [(kh*3+kw)*32+ci][n]
    else         v = krec[(size_t)(k - 288) * G4 + n];   // [(kh*3+kw)*64+ci][n]
    wp[(size_t)n * KTOT + k] = (_Float16)v;
}

// Zero padded h buffer (incl. halo) and c state.
__global__ void init_state(float* __restrict__ c, _Float16* __restrict__ h) {
    int i = blockIdx.x * 256 + threadIdx.x;
    const int nh = BB * HP * WP * COUT;
    const int nc = BB * HH * WW * COUT;
    if (i < nh) h[i] = (_Float16)0.0f;
    if (i < nc) c[i] = 0.0f;
}

// ---------------- fragment loaders ----------------
// A layout (16-bit 16x32): lanes 0-15 hold K {0..7,16..23}, lanes 16-31 hold K {8..15,24..31}
__device__ __forceinline__ v16h load_a_frag(const _Float16* __restrict__ p) {
    union { v16h v; h8 h[2]; } A;
    A.h[0] = *(const h8*)(p);
    A.h[1] = *(const h8*)(p + 16);
    return A.v;
}

// B layout (16-bit 32x16): lane = column, K {0..7,16..23} (kh2=0) / {8..15,24..31} (kh2=1)
__device__ __forceinline__ v16h load_b_frag_lds(const _Float16* wk) {
    union { v16h v; h8 h[2]; } B;
    B.h[0] = *(const h8*)(wk);
    B.h[1] = *(const h8*)(wk + 16);
    return B.v;
}

// ---------------- fused gate GEMM (implicit im2col, WMMA f16 -> f32) ----------------
// Block: 8 waves sharing one 64-channel gate group; weights staged in LDS (~109 KB).
// LDS layout: channel n = j*16+n16, K k  ->  halfs offset n16*LDSG + j*LDSJ + k
//   (per-lane base covers n16/kh2; j*LDSJ + kc*32 is always a DS immediate offset)
// Wave: 32 pixels x 64 gate channels (8 accumulators), fully unrolled K = 864.
// g[b,y,x, 0:256] = conv(x_t, kernel) + conv(h, rec_kernel) + bias

__global__ __launch_bounds__(256)
void convlstm_gates_wmma(const _Float16* __restrict__ x16,   // (B,T,HP,WP,CIN) padded
                         const _Float16* __restrict__ h16,   // (B,HP,WP,COUT) padded
                         const _Float16* __restrict__ wp,    // (G4, KTOT)
                         const float*    __restrict__ bias,  // (G4)
                         float*          __restrict__ g,     // (B,H,W,G4)
                         int t)
{
    extern __shared__ _Float16 wlds[];                 // 16 groups x LDSG halfs

    const int tid = threadIdx.x;
    int bid = blockIdx.x;                              // 256 blocks
    const int nt4  = bid & 3;  bid >>= 2;              // gate-channel group (64 ch)
    const int ygrp = bid & 15; bid >>= 4;              // group of 4 rows
    const int b    = bid;                              // 0..3

    // ---- stage this block's 64x864 weight tile into LDS (permuted layout) ----
    const _Float16* wsrc = wp + (size_t)(nt4 * 64) * KTOT;
    for (int idx = tid; idx < 64 * 108; idx += 256) {  // 108 groups of 8 halfs per row
        const int n  = idx / 108;
        const int kg = idx % 108;
        *(h8*)&wlds[(n & 15) * LDSG + (n >> 4) * LDSJ + kg * 8] =
            *(const h8*)&wsrc[(size_t)n * KTOT + kg * 8];
    }
    __syncthreads();

    const int lane = tid & 31;
    const int wave = tid >> 5;
    const int yoff = wave >> 1;                        // 0..3
    const int xt2  = wave & 1;                         // 32-pixel tile along W
    const int y    = ygrp * 4 + yoff;

    const int m   = lane & 15;
    const int kh2 = lane >> 4;
    const int px0 = xt2 * 32 + m;                      // pixel for M-tile 0 (tile 1 = +16)

    v8f acc0[4] = {};                                  // M-tile 0, 4 N-tiles
    v8f acc1[4] = {};                                  // M-tile 1, 4 N-tiles

    // Center-pixel base pointers into padded images; all tap offsets are immediates.
    const _Float16* ax0 = x16 + ((size_t)(b * TT + t) * HP * WP
                               + (size_t)(y + 1) * WP + (px0 + 1)) * CIN + kh2 * 8;
    const _Float16* ah0 = h16 + ((size_t)b * HP * WP
                               + (size_t)(y + 1) * WP + (px0 + 1)) * COUT + kh2 * 8;
    const _Float16* wrow = &wlds[m * LDSG + kh2 * 8]; // column m, K-half kh2

    // ---- input-conv taps: chunks 0..8, 32 input channels each
    #pragma unroll
    for (int tap = 0; tap < 9; ++tap) {
        const int dy = tap / 3 - 1, dx = tap % 3 - 1;
        const int off = (dy * WP + dx) * CIN;                 // compile-time constant
        const v16h A0 = load_a_frag(ax0 + off);
        const v16h A1 = load_a_frag(ax0 + off + 16 * CIN);    // +16 pixels
        v16h Bf[4];
        #pragma unroll
        for (int j = 0; j < 4; ++j) Bf[j] = load_b_frag_lds(wrow + j * LDSJ + tap * 32);
        #pragma unroll
        for (int j = 0; j < 4; ++j) {
            acc0[j] = WMMA_F16(A0, Bf[j], acc0[j]);
            acc1[j] = WMMA_F16(A1, Bf[j], acc1[j]);
        }
    }

    // ---- recurrent-conv taps: chunks 9..26, 2 chunks of 32 channels per tap
    #pragma unroll
    for (int r = 0; r < 18; ++r) {
        const int tap = r >> 1;
        const int cb  = (r & 1) * 32;
        const int dy = tap / 3 - 1, dx = tap % 3 - 1;
        const int off = (dy * WP + dx) * COUT + cb;           // compile-time constant
        const v16h A0 = load_a_frag(ah0 + off);
        const v16h A1 = load_a_frag(ah0 + off + 16 * COUT);   // +16 pixels
        v16h Bf[4];
        #pragma unroll
        for (int j = 0; j < 4; ++j) Bf[j] = load_b_frag_lds(wrow + j * LDSJ + (9 + r) * 32);
        #pragma unroll
        for (int j = 0; j < 4; ++j) {
            acc0[j] = WMMA_F16(A0, Bf[j], acc0[j]);
            acc1[j] = WMMA_F16(A1, Bf[j], acc1[j]);
        }
    }

    // ---- store D + bias: VGPR r -> M = r + 8*kh2 (+16 for tile 1), N = lane&15
    float* gout = g + ((size_t)(b * HH + y) * WW) * G4;
    const int chb = nt4 * 64 + m;
    float bv[4];
    #pragma unroll
    for (int j = 0; j < 4; ++j) bv[j] = bias[chb + j * 16];

    #pragma unroll
    for (int r = 0; r < 8; ++r) {
        const int po0 = xt2 * 32 + kh2 * 8 + r;
        float* q0 = gout + (size_t)po0 * G4 + chb;
        float* q1 = q0 + (size_t)16 * G4;
        #pragma unroll
        for (int j = 0; j < 4; ++j) {
            q0[j * 16] = acc0[j][r] + bv[j];
            q1[j * 16] = acc1[j][r] + bv[j];
        }
    }
}

// ---------------- pointwise gates + state update + BN ----------------

__device__ __forceinline__ float hsig(float z) {
    return fminf(fmaxf(0.2f * z + 0.5f, 0.0f), 1.0f);
}

__global__ __launch_bounds__(256)
void lstm_pointwise(const float* __restrict__ g, float* __restrict__ cbuf,
                    _Float16* __restrict__ h16, float* __restrict__ yout,
                    const float* __restrict__ gamma, const float* __restrict__ beta,
                    const float* __restrict__ mmean, const float* __restrict__ mvar,
                    int t)
{
    int i = blockIdx.x * 256 + threadIdx.x;          // over B*H*W*COUT
    if (i >= BB * HH * WW * COUT) return;
    const int ch  = i & 63;
    const int pix = i >> 6;                          // b*4096 + s
    const int b   = pix >> 12;
    const int s   = pix & 4095;
    const int yy  = s >> 6;
    const int xx  = s & 63;

    const float* gp = g + (size_t)pix * G4 + ch;
    const float gi = gp[0], gf = gp[64], gc = gp[128], go = gp[192];

    const float cold = cbuf[i];
    const float cn = hsig(gf) * cold + hsig(gi) * tanhf(gc);
    const float h  = hsig(go) * tanhf(cn);

    cbuf[i] = cn;
    // write h into interior of padded buffer (halo stays zero)
    h16[((size_t)b * HP * WP + (size_t)(yy + 1) * WP + (xx + 1)) * COUT + ch] = (_Float16)h;

    const float inv = gamma[ch] * rsqrtf(mvar[ch] + 1e-3f);
    yout[((size_t)(b * TT + t) * 4096 + s) * COUT + ch] = h * inv + (beta[ch] - mmean[ch] * inv);
}

// ---------------- host ----------------

extern "C" void kernel_launch(void* const* d_in, const int* in_sizes, int n_in,
                              void* d_out, int out_size, void* d_ws, size_t ws_size,
                              hipStream_t stream) {
    const float* x     = (const float*)d_in[0];
    const float* kin   = (const float*)d_in[1];
    const float* krec  = (const float*)d_in[2];
    const float* bias  = (const float*)d_in[3];
    const float* gamma = (const float*)d_in[4];
    const float* beta  = (const float*)d_in[5];
    const float* mmean = (const float*)d_in[6];
    const float* mvar  = (const float*)d_in[7];
    float* out = (float*)d_out;

    char* ws = (char*)d_ws;
    size_t off = 0;
    auto alloc = [&](size_t bytes) -> void* {
        void* p = ws + off;
        off = (off + bytes + 255) & ~(size_t)255;
        return p;
    };
    _Float16* x16 = (_Float16*)alloc((size_t)BB * TT * HP * WP * CIN * 2);
    _Float16* wp  = (_Float16*)alloc((size_t)G4 * KTOT * 2);
    _Float16* h16 = (_Float16*)alloc((size_t)BB * HP * WP * COUT * 2);
    float*    cbf = (float*)   alloc((size_t)BB * HH * WW * COUT * 4);
    float*    g   = (float*)   alloc((size_t)BB * HH * WW * G4   * 4);

    const int nxp = BB * TT * HP * WP * CIN;
    pad_x_f16<<<(nxp + 255) / 256, 256, 0, stream>>>(x, x16);

    const int nw = G4 * KTOT;
    pack_weights<<<(nw + 255) / 256, 256, 0, stream>>>(kin, krec, wp);

    const int nh = BB * HP * WP * COUT;
    init_state<<<(nh + 255) / 256, 256, 0, stream>>>(cbf, h16);

    const size_t lds_bytes = (size_t)16 * LDSG * 2;   // 111,872 B
    const int ns = BB * HH * WW * COUT;
    for (int t = 0; t < TT; ++t) {
        convlstm_gates_wmma<<<256, 256, lds_bytes, stream>>>(x16, h16, wp, bias, g, t);
        lstm_pointwise<<<(ns + 255) / 256, 256, 0, stream>>>(g, cbf, h16, out,
                                                             gamma, beta, mmean, mvar, t);
    }
}
